// GatedDeltaNet_41016937677096
// MI455X (gfx1250) — matrix operand order repaired and checked by reading, not compiled, not run on hardware
//
#include <hip/hip_runtime.h>
#include <hip/hip_bf16.h>
#include <math.h>

// ---------------- problem constants ----------------
namespace {
constexpr int kS    = 4096;
constexpr int kHID  = 2048;
constexpr int kHV   = 32;
constexpr int kDK   = 128;
constexpr int kDV   = 128;
constexpr int kCH   = 64;     // chunk length
constexpr int kN    = 64;     // number of chunks = S / CH
constexpr int kKEY  = 2048;   // HK * DK
constexpr int kVAL  = 4096;   // HV * DV
constexpr int kCONV = 8192;   // 2*KEY + VAL
}

typedef __attribute__((ext_vector_type(16))) __bf16 v16bf;
typedef __attribute__((ext_vector_type(8)))  float  v8f;

// ---------------- WMMA helpers ----------------
__device__ __forceinline__ v8f zero8() {
  v8f z;
#pragma unroll
  for (int e = 0; e < 8; ++e) z[e] = 0.f;
  return z;
}

__device__ __forceinline__ v8f wmma_bf16(v16bf a, v16bf b, v8f c) {
  return __builtin_amdgcn_wmma_f32_16x16x32_bf16(false, a, false, b, (short)0, c,
                                                 false, false);
}

// Async global -> LDS copy, 16 bytes per lane (tracked by ASYNCcnt).
__device__ __forceinline__ void async_copy_b128(const float* g, unsigned lds) {
  asm volatile("global_load_async_to_lds_b128 %0, %1, off"
               :: "v"(lds), "v"(g)
               : "memory");
}
__device__ __forceinline__ void wait_async(void) {
  asm volatile("s_wait_asynccnt 0x0" ::: "memory");
}
__device__ __forceinline__ void wait_async6(void) {
  asm volatile("s_wait_asynccnt 0x6" ::: "memory");
}

// A operand fragment (16x32 tile, MxK) from row-major f32, with per-row scale.
// Layout (ISA 7.12.2, 16-bit A 16x32): lanes 0-15 row M=lane, K={0..7,16..23};
// lanes 16-31 same rows, K={8..15,24..31}.
__device__ __forceinline__ v16bf fragA_f32(const float* __restrict__ base, int ld,
                                           int row0, int col0, float rs) {
  const int lane = threadIdx.x & 31;
  const int r = lane & 15, hi = lane >> 4;
  const float* p = base + (size_t)(row0 + r) * ld + col0 + hi * 8;
  v16bf f;
#pragma unroll
  for (int e = 0; e < 8; ++e) f[e] = (__bf16)(p[e] * rs);
#pragma unroll
  for (int e = 0; e < 8; ++e) f[8 + e] = (__bf16)(p[16 + e] * rs);
  return f;
}

__device__ __forceinline__ v16bf fragA_lds(const __bf16* __restrict__ base, int ld,
                                           int row0, int col0) {
  const int lane = threadIdx.x & 31;
  const int r = lane & 15, hi = lane >> 4;
  const __bf16* p = base + (size_t)(row0 + r) * ld + col0 + hi * 8;
  v16bf f;
#pragma unroll
  for (int e = 0; e < 8; ++e) f[e] = p[e];
#pragma unroll
  for (int e = 0; e < 8; ++e) f[8 + e] = p[16 + e];
  return f;
}

// B operand fragment (32x16 tile, KxN) sourced from [N][K] row-major storage.
// Layout: lane L (0-15) column N=L holds K={0..7,16..23}; lane L+16 K={8..15,24..31}.
__device__ __forceinline__ v16bf fragB_f32(const float* __restrict__ base, int ld,
                                           int n0, int col0) {
  const int lane = threadIdx.x & 31;
  const int r = lane & 15, hi = lane >> 4;
  const float* p = base + (size_t)(n0 + r) * ld + col0 + hi * 8;
  v16bf f;
#pragma unroll
  for (int e = 0; e < 16; ++e) f[e] = (__bf16)p[e < 8 ? e : e + 8];
  return f;
}

__device__ __forceinline__ v16bf fragB_lds(const __bf16* __restrict__ base, int ld,
                                           int n0, int col0) {
  const int lane = threadIdx.x & 31;
  const int r = lane & 15, hi = lane >> 4;
  const __bf16* p = base + (size_t)(n0 + r) * ld + col0 + hi * 8;
  v16bf f;
#pragma unroll
  for (int e = 0; e < 16; ++e) f[e] = p[e < 8 ? e : e + 8];
  return f;
}

// ---------------- kernel 1: generic GEMM  C[M,N] = A[M,K] * B[N,K]^T ----------
// 256 threads = 8 waves; block tile 128x64; k-step 32; bf16 WMMA, f32 accum.
// Tiles staged via async global->LDS copies, double buffered.
__global__ __launch_bounds__(256) void gemm_nt_bf16(
    const float* __restrict__ A, const float* __restrict__ B,
    float* __restrict__ C, int M, int N, int K) {
  __shared__ float Asf[2][128 * 32];           // 2 x 16 KB
  __shared__ float Bsf[2][64 * 32];            // 2 x 8 KB
  __shared__ __align__(16) float Bdump[8];     // sink for OOB-row async copies
  const int tid = threadIdx.x;
  const int bm = blockIdx.y * 128;
  const int bn = blockIdx.x * 64;
  const int wave = tid >> 5;
  const int wr = wave >> 1;  // 0..3
  const int wc = wave & 1;   // 0..1

  // rows of B beyond N contribute zeros; fill once (never overwritten: their
  // async copies are redirected to Bdump below).
  {
    int r = tid >> 2, c0 = (tid & 3) * 8;
    if (bn + r >= N) {
#pragma unroll
      for (int j = 0; j < 8; ++j) {
        Bsf[0][r * 32 + c0 + j] = 0.f;
        Bsf[1][r * 32 + c0 + j] = 0.f;
      }
    }
  }

  // issue one 128x32 A tile + 64x32 B tile: exactly 6 async ops per lane
  auto issue_tile = [&](int k0, int buf) {
    {
      int r = tid >> 1, c0 = (tid & 1) * 16;  // 16 floats = 4 x b128
      const float* src = A + (size_t)(bm + r) * K + k0 + c0;
      __builtin_prefetch(src + 64, 0, 1);  // tile after next
      unsigned dst = (unsigned)(uintptr_t)&Asf[buf][r * 32 + c0];
#pragma unroll
      for (int j = 0; j < 4; ++j) async_copy_b128(src + 4 * j, dst + 16 * j);
    }
    {
      int r = tid >> 2, c0 = (tid & 3) * 8;  // 8 floats = 2 x b128
      int gn = bn + r;
      bool ok = gn < N;
      const float* src = B + (size_t)(ok ? gn : 0) * K + k0 + c0;
      unsigned dst = ok ? (unsigned)(uintptr_t)&Bsf[buf][r * 32 + c0]
                        : (unsigned)(uintptr_t)&Bdump[0];
      async_copy_b128(src, dst);
      async_copy_b128(src + 4, dst + 16);
    }
  };

  v8f acc[2][2];
#pragma unroll
  for (int i = 0; i < 2; ++i)
#pragma unroll
    for (int j = 0; j < 2; ++j) acc[i][j] = zero8();

  const int nk = K / 32;
  issue_tile(0, 0);
  for (int ks = 0; ks < nk; ++ks) {
    const int buf = ks & 1;
    if (ks + 1 < nk) {
      issue_tile((ks + 1) * 32, buf ^ 1);
      wait_async6();  // 6 newer ops may remain in flight; tile `ks` has landed
    } else {
      wait_async();
    }
    __syncthreads();
#pragma unroll
    for (int i = 0; i < 2; ++i) {
      v16bf a = fragA_f32(&Asf[buf][0], 32, wr * 32 + i * 16, 0, 1.f);
#pragma unroll
      for (int j = 0; j < 2; ++j) {
        v16bf b = fragB_f32(&Bsf[buf][0], 32, wc * 32 + j * 16, 0);
        acc[i][j] = wmma_bf16(a, b, acc[i][j]);
      }
    }
    __syncthreads();
  }
  const int lane = tid & 31;
  const int l15 = lane & 15, mo = (lane >> 4) * 8;
#pragma unroll
  for (int i = 0; i < 2; ++i)
#pragma unroll
    for (int j = 0; j < 2; ++j)
#pragma unroll
      for (int vv = 0; vv < 8; ++vv) {
        int rr = bm + wr * 32 + i * 16 + mo + vv;
        int cc = bn + wc * 32 + j * 16 + l15;
        if (rr < M && cc < N) C[(size_t)rr * N + cc] = acc[i][j][vv];
      }
}

// ---------------- kernel 2: causal conv (K=4) + SiLU ----------------
__global__ __launch_bounds__(256) void conv_silu_kernel(
    const float* __restrict__ qkv, const float* __restrict__ conv_w,
    float* __restrict__ mixed) {
  size_t idx = (size_t)blockIdx.x * 256 + threadIdx.x;  // over S*CONV
  int c = (int)(idx % kCONV);
  int s = (int)(idx / kCONV);
  float acc = conv_w[c * 4 + 3] * qkv[idx];
  if (s >= 1) acc += conv_w[c * 4 + 2] * qkv[idx - kCONV];
  if (s >= 2) acc += conv_w[c * 4 + 1] * qkv[idx - 2 * kCONV];
  if (s >= 3) acc += conv_w[c * 4 + 0] * qkv[idx - 3 * kCONV];
  mixed[idx] = acc / (1.f + __expf(-acc));
}

// ---------------- kernel 3: beta / g gates ----------------
__global__ __launch_bounds__(256) void gate_kernel(
    const float* __restrict__ b, const float* __restrict__ a,
    const float* __restrict__ A_log, const float* __restrict__ dt_bias,
    float* __restrict__ beta, float* __restrict__ g) {
  int idx = blockIdx.x * 256 + threadIdx.x;  // over S*HV
  int h = idx & (kHV - 1);
  beta[idx] = 1.f / (1.f + __expf(-b[idx]));
  float x = a[idx] + dt_bias[h];
  float sp = (x > 20.f) ? x : log1pf(__expf(x));
  g[idx] = -__expf(A_log[h]) * sp;
}

// ---------------- kernel 4: per-(head,chunk) cumsum of g ----------------
__global__ __launch_bounds__(256) void gcumsum_kernel(const float* __restrict__ g,
                                                      float* __restrict__ gc) {
  int idx = blockIdx.x * 256 + threadIdx.x;  // HV * N
  if (idx >= kHV * kN) return;
  int h = idx / kN, n = idx % kN;
  int s0 = n * kCH;
  float acc = 0.f;
  for (int i = 0; i < kCH; ++i) {
    acc += g[(s0 + i) * kHV + h];
    gc[h * kS + s0 + i] = acc;
  }
}

// ---------------- kernel 5: q/k l2norm + head repeat ----------------
__global__ __launch_bounds__(256) void qknorm_kernel(
    const float* __restrict__ mixed, float* __restrict__ qh,
    float* __restrict__ kh) {
  int gw = blockIdx.x * 8 + (threadIdx.x >> 5);  // wave over S*HV
  int lane = threadIdx.x & 31;
  int h = gw & (kHV - 1);
  int s = gw >> 5;
  int khead = h >> 1;  // repeat factor 2
  const float* qrow = mixed + (size_t)s * kCONV + khead * kDK;
  const float* krow = qrow + kKEY;
  float q[4], k[4], sq = 0.f, sk = 0.f;
#pragma unroll
  for (int t = 0; t < 4; ++t) {
    q[t] = qrow[lane * 4 + t]; sq += q[t] * q[t];
    k[t] = krow[lane * 4 + t]; sk += k[t] * k[t];
  }
#pragma unroll
  for (int m = 16; m >= 1; m >>= 1) {
    sq += __shfl_xor(sq, m, 32);
    sk += __shfl_xor(sk, m, 32);
  }
  float rq = rsqrtf(sq + 1e-6f) * 0.08838834764831845f;  // * DK^-0.5
  float rk = rsqrtf(sk + 1e-6f);
  float* qd = qh + ((size_t)h * kS + s) * kDK + lane * 4;
  float* kd = kh + ((size_t)h * kS + s) * kDK + lane * 4;
#pragma unroll
  for (int t = 0; t < 4; ++t) { qd[t] = q[t] * rq; kd[t] = k[t] * rk; }
}

// ---------------- kernel 6: chunk prep (A0, T-solve, u, w) ----------------
// one block per (head, chunk): blockIdx.x = h*64 + n
__global__ __launch_bounds__(256) void chunk_prep_kernel(
    const float* __restrict__ mixed, const float* __restrict__ kh,
    const float* __restrict__ beta, const float* __restrict__ gc,
    float* __restrict__ u_g, float* __restrict__ w_g) {
  __shared__ float  sA0[kCH * kCH];         // 16 KB
  __shared__ float  sTf[kCH * (kCH + 1)];   // 16.25 KB
  __shared__ __bf16 sT[kCH * kCH];          // 8 KB
  __shared__ float  sGC[kCH], sBeta[kCH], sKG[kCH];
  const int h = blockIdx.x >> 6;
  const int n = blockIdx.x & 63;
  const int s0 = n * kCH;
  const int tid = threadIdx.x, lane = tid & 31, wave = tid >> 5;
  const int l15 = lane & 15, hi = lane >> 4, mo = hi * 8;

  if (tid < kCH) {
    float gcv = gc[h * kS + s0 + tid];
    float bt = beta[(s0 + tid) * kHV + h];
    sGC[tid] = gcv; sBeta[tid] = bt; sKG[tid] = bt * __expf(gcv);
  }
  __syncthreads();

  const float* kbase = kh + ((size_t)h * kS + s0) * kDK;

  // A0 = -(k_beta @ k^T) * decay, strictly lower   (16 tiles / 8 waves)
  for (int t = wave; t < 16; t += 8) {
    int ti = t >> 2, tj = t & 3;
    v8f acc = zero8();
    float rs = sBeta[ti * 16 + l15];
#pragma unroll
    for (int ks = 0; ks < 4; ++ks) {
      v16bf a = fragA_f32(kbase, kDK, ti * 16, ks * 32, rs);
      v16bf b = fragB_f32(kbase, kDK, tj * 16, ks * 32);
      acc = wmma_bf16(a, b, acc);
    }
#pragma unroll
    for (int vv = 0; vv < 8; ++vv) {
      int i = ti * 16 + mo + vv, j = tj * 16 + l15;
      sA0[i * kCH + j] = (i > j) ? -acc[vv] * __expf(sGC[i] - sGC[j]) : 0.f;
    }
  }
  __syncthreads();

  // T = (I - A0)^-1 via forward substitution; thread j owns column j
  if (tid < kCH) {
    int j = tid;
    for (int i = 0; i < kCH; ++i) {
      float t = (i == j) ? 1.f : 0.f;
      for (int m = j; m < i; ++m) t += sA0[i * kCH + m] * sTf[m * (kCH + 1) + j];
      sTf[i * (kCH + 1) + j] = t;
    }
    for (int i = 0; i < kCH; ++i) sT[i * kCH + j] = (__bf16)sTf[i * (kCH + 1) + j];
  }
  __syncthreads();

  // u = T @ v_beta ; w = T @ (k_beta * exp(gc))   (64 tiles / 8 waves)
  const float* vbase = mixed + (size_t)s0 * kCONV + 2 * kKEY + h * kDV;
  for (int t = wave; t < 64; t += 8) {
    bool isU = t < 32;
    int tt = isU ? t : t - 32;
    int ti = tt >> 3, tj = tt & 7;
    v8f acc = zero8();
    int d = tj * 16 + l15;
#pragma unroll
    for (int ks = 0; ks < 2; ++ks) {
      int k0 = ks * 32;
      v16bf a = fragA_lds(sT, kCH, ti * 16, k0);
      v16bf b;
      if (isU) {
#pragma unroll
        for (int e = 0; e < 16; ++e) {
          int j = k0 + hi * 8 + (e < 8 ? e : e + 8);
          b[e] = (__bf16)(vbase[(size_t)j * kCONV + d] * sBeta[j]);
        }
      } else {
#pragma unroll
        for (int e = 0; e < 16; ++e) {
          int j = k0 + hi * 8 + (e < 8 ? e : e + 8);
          b[e] = (__bf16)(kbase[(size_t)j * kDK + d] * sKG[j]);
        }
      }
      acc = wmma_bf16(a, b, acc);
    }
    float* dst = (isU ? u_g : w_g) + ((size_t)h * kS + s0) * kDK;
#pragma unroll
    for (int vv = 0; vv < 8; ++vv) {
      int i = ti * 16 + mo + vv;
      dst[(size_t)i * kDK + tj * 16 + l15] = acc[vv];
    }
  }
}

// ---------------- kernel 7: sequential chunk scan ----------------
// one block per head; 128x128 f32 state lives in WMMA accumulators (8 v8f/wave)
__global__ __launch_bounds__(256) void scan_kernel(
    const float* __restrict__ qh, const float* __restrict__ kh,
    const float* __restrict__ u_g, const float* __restrict__ w_g,
    const float* __restrict__ gc, float* __restrict__ core) {
  __shared__ __bf16 stateT[kDV * kDK];  // [v][d] 32 KB
  __shared__ __bf16 sVNT[kDV * kCH];    // [v][i] 16 KB
  __shared__ __bf16 sAttn[kCH * kCH];   // 8 KB
  __shared__ float sGC[kCH], sEXG[kCH], sKD[kCH];
  const int h = blockIdx.x;
  const int tid = threadIdx.x, lane = tid & 31, wave = tid >> 5;
  const int l15 = lane & 15, hi = lane >> 4, mo = hi * 8;

  for (int i = tid; i < kDV * kDK; i += 256) stateT[i] = (__bf16)0.f;
  v8f st[8];
#pragma unroll
  for (int j = 0; j < 8; ++j) st[j] = zero8();
  __syncthreads();

  const int ti = wave >> 1;
  const int tj0 = (wave & 1) * 4;

  for (int n = 0; n < kN; ++n) {
    const int s0 = n * kCH;
    if (tid < kCH) {
      float gv = gc[h * kS + s0 + tid];
      sGC[tid] = gv;
      sEXG[tid] = __expf(gv);
    }
    __syncthreads();
    const float gl = sGC[kCH - 1];
    if (tid < kCH) sKD[tid] = __expf(gl - sGC[tid]);  // consumed after next barrier

    const float* wb = w_g + ((size_t)h * kS + s0) * kDK;
    const float* ub = u_g + ((size_t)h * kS + s0) * kDV;
    const float* qb = qh + ((size_t)h * kS + s0) * kDK;
    const float* kb = kh + ((size_t)h * kS + s0) * kDK;

    // v_new = u - w @ state     (4 tiles per wave)
    {
      v8f acc[4];
#pragma unroll
      for (int jj = 0; jj < 4; ++jj) acc[jj] = zero8();
#pragma unroll
      for (int ks = 0; ks < 4; ++ks) {
        v16bf a = fragA_f32(wb, kDK, ti * 16, ks * 32, 1.f);
#pragma unroll
        for (int jj = 0; jj < 4; ++jj)
          acc[jj] = wmma_bf16(a, fragB_lds(stateT, kDK, (tj0 + jj) * 16, ks * 32),
                              acc[jj]);
      }
#pragma unroll
      for (int jj = 0; jj < 4; ++jj)
#pragma unroll
        for (int vv = 0; vv < 8; ++vv) {
          int i = ti * 16 + mo + vv, v = (tj0 + jj) * 16 + l15;
          float val = ub[(size_t)i * kDV + v] - acc[jj][vv];
          sVNT[v * kCH + i] = (__bf16)val;
        }
    }
    // attn = (q @ k^T) * decay * maskL   (2 tiles per wave)
#pragma unroll
    for (int q2 = 0; q2 < 2; ++q2) {
      int tj = (wave & 1) * 2 + q2;
      v8f acc = zero8();
#pragma unroll
      for (int ks = 0; ks < 4; ++ks)
        acc = wmma_bf16(fragA_f32(qb, kDK, ti * 16, ks * 32, 1.f),
                        fragB_f32(kb, kDK, tj * 16, ks * 32), acc);
#pragma unroll
      for (int vv = 0; vv < 8; ++vv) {
        int i = ti * 16 + mo + vv, j = tj * 16 + l15;
        float val = (i >= j) ? acc[vv] * __expf(sGC[i] - sGC[j]) : 0.f;
        sAttn[i * kCH + j] = (__bf16)val;
      }
    }
    __syncthreads();

    // out = (q * exp(gc)) @ state + attn @ v_new
    {
      v8f acc[4];
#pragma unroll
      for (int jj = 0; jj < 4; ++jj) acc[jj] = zero8();
      float rs = sEXG[ti * 16 + l15];
#pragma unroll
      for (int ks = 0; ks < 4; ++ks) {
        v16bf a = fragA_f32(qb, kDK, ti * 16, ks * 32, rs);
#pragma unroll
        for (int jj = 0; jj < 4; ++jj)
          acc[jj] = wmma_bf16(a, fragB_lds(stateT, kDK, (tj0 + jj) * 16, ks * 32),
                              acc[jj]);
      }
#pragma unroll
      for (int ks = 0; ks < 2; ++ks) {
        v16bf a = fragA_lds(sAttn, kCH, ti * 16, ks * 32);
#pragma unroll
        for (int jj = 0; jj < 4; ++jj)
          acc[jj] = wmma_bf16(a, fragB_lds(sVNT, kCH, (tj0 + jj) * 16, ks * 32),
                              acc[jj]);
      }
      float* ob = core + ((size_t)h * kS + s0) * kDV;
#pragma unroll
      for (int jj = 0; jj < 4; ++jj)
#pragma unroll
        for (int vv = 0; vv < 8; ++vv) {
          int i = ti * 16 + mo + vv, v = (tj0 + jj) * 16 + l15;
          ob[(size_t)i * kDV + v] = acc[jj][vv];
        }
    }
    // state = state * exp(gl) + kd^T @ v_new   (wave owns d-rows wave*16..+15)
    {
      float egl = __expf(gl);
#pragma unroll
      for (int j = 0; j < 8; ++j)
#pragma unroll
        for (int e = 0; e < 8; ++e) st[j][e] *= egl;
      int d = wave * 16 + l15;
#pragma unroll
      for (int ks = 0; ks < 2; ++ks) {
        int k0 = ks * 32;
        v16bf a;  // A[d][i] = k[i][d] * exp(gl - gc[i])
#pragma unroll
        for (int e = 0; e < 16; ++e) {
          int i = k0 + ((e < 8) ? (e + mo) : (e + 8 + mo));
          a[e] = (__bf16)(kb[(size_t)i * kDK + d] * sKD[i]);
        }
#pragma unroll
        for (int j = 0; j < 8; ++j)
          st[j] = wmma_bf16(a, fragB_lds(sVNT, kCH, j * 16, k0), st[j]);
      }
      __syncthreads();  // all readers of old stateT done
#pragma unroll
      for (int j = 0; j < 8; ++j)
#pragma unroll
        for (int vv = 0; vv < 8; ++vv) {
          int dd = wave * 16 + mo + vv, v = j * 16 + l15;
          stateT[v * kDK + dd] = (__bf16)st[j][vv];
        }
      __syncthreads();
    }
  }
}

// ---------------- kernel 8: SiLU gate + RMSNorm ----------------
__global__ __launch_bounds__(256) void gatenorm_kernel(
    const float* __restrict__ core, const float* __restrict__ z,
    const float* __restrict__ norm_w, float* __restrict__ xnorm) {
  int gw = blockIdx.x * 8 + (threadIdx.x >> 5);  // wave over S*HV rows
  int lane = threadIdx.x & 31;
  int h = gw & (kHV - 1);
  int s = gw >> 5;
  const float* cb = core + ((size_t)h * kS + s) * kDV + lane * 4;
  const float* zb = z + (size_t)s * kVAL + h * kDV + lane * 4;
  float xg[4], ss = 0.f;
#pragma unroll
  for (int t = 0; t < 4; ++t) {
    float zz = zb[t];
    float sz = zz / (1.f + __expf(-zz));
    xg[t] = cb[t] * sz;
    ss += xg[t] * xg[t];
  }
#pragma unroll
  for (int m = 16; m >= 1; m >>= 1) ss += __shfl_xor(ss, m, 32);
  float sc = rsqrtf(ss * (1.f / 128.f) + 1e-6f);
  float* ob = xnorm + (size_t)s * kVAL + h * kDV + lane * 4;
#pragma unroll
  for (int t = 0; t < 4; ++t) ob[t] = xg[t] * sc * norm_w[lane * 4 + t];
}

// ---------------- host side ----------------
extern "C" void kernel_launch(void* const* d_in, const int* in_sizes, int n_in,
                              void* d_out, int out_size, void* d_ws, size_t ws_size,
                              hipStream_t stream) {
  (void)in_sizes; (void)n_in; (void)out_size; (void)ws_size;
  const float* h       = (const float*)d_in[0];
  const float* W_qkv   = (const float*)d_in[1];
  const float* W_z     = (const float*)d_in[2];
  const float* W_b     = (const float*)d_in[3];
  const float* W_a     = (const float*)d_in[4];
  const float* conv_w  = (const float*)d_in[5];
  const float* A_log   = (const float*)d_in[6];
  const float* dt_bias = (const float*)d_in[7];
  const float* norm_w  = (const float*)d_in[8];
  const float* W_out   = (const float*)d_in[9];
  float* out = (float*)d_out;

  float* ws = (float*)d_ws;
  size_t off = 0;
  auto alloc = [&](size_t nelem) { float* p = ws + off; off += nelem; return p; };
  float* qkv   = alloc((size_t)kS * kCONV);   // reused for core+xnorm later
  float* mixed = alloc((size_t)kS * kCONV);
  float* zbuf  = alloc((size_t)kS * kVAL);
  float* bbuf  = alloc((size_t)kS * kHV);
  float* abuf  = alloc((size_t)kS * kHV);
  float* beta  = alloc((size_t)kS * kHV);
  float* gbuf  = alloc((size_t)kS * kHV);
  float* gcg   = alloc((size_t)kHV * kS);
  float* qhb   = alloc((size_t)kHV * kS * kDK);
  float* khb   = alloc((size_t)kHV * kS * kDK);
  float* ubuf  = alloc((size_t)kHV * kS * kDV);
  float* wkb   = alloc((size_t)kHV * kS * kDK);
  float* coreb = qkv;                          // qkv dead after conv
  float* xnorm = qkv + (size_t)kHV * kS * kDV; // second half of qkv region

  dim3 blk(256);
  gemm_nt_bf16<<<dim3(kCONV / 64, kS / 128), blk, 0, stream>>>(h, W_qkv, qkv, kS, kCONV, kHID);
  gemm_nt_bf16<<<dim3(kVAL / 64, kS / 128), blk, 0, stream>>>(h, W_z, zbuf, kS, kVAL, kHID);
  gemm_nt_bf16<<<dim3(1, kS / 128), blk, 0, stream>>>(h, W_b, bbuf, kS, kHV, kHID);
  gemm_nt_bf16<<<dim3(1, kS / 128), blk, 0, stream>>>(h, W_a, abuf, kS, kHV, kHID);
  conv_silu_kernel<<<(unsigned)((size_t)kS * kCONV / 256), blk, 0, stream>>>(qkv, conv_w, mixed);
  gate_kernel<<<kS * kHV / 256, blk, 0, stream>>>(bbuf, abuf, A_log, dt_bias, beta, gbuf);
  gcumsum_kernel<<<(kHV * kN + 255) / 256, blk, 0, stream>>>(gbuf, gcg);
  qknorm_kernel<<<kS * kHV / 8, blk, 0, stream>>>(mixed, qhb, khb);
  chunk_prep_kernel<<<kHV * kN, blk, 0, stream>>>(mixed, khb, beta, gcg, ubuf, wkb);
  scan_kernel<<<kHV, blk, 0, stream>>>(qhb, khb, ubuf, wkb, gcg, coreb);
  gatenorm_kernel<<<kS * kHV / 8, blk, 0, stream>>>(coreb, zbuf, norm_w, xnorm);
  gemm_nt_bf16<<<dim3(kHID / 64, kS / 128), blk, 0, stream>>>(xnorm, W_out, out, kS, kHID, kVAL);
}
